// GAT_23862838297157
// MI455X (gfx1250) — matrix operand order repaired and checked by reading, not compiled, not run on hardware
//
#include <hip/hip_runtime.h>

// ---------------------------------------------------------------------------
// GAT (3 GATConv layers + MLP head) for MI455X / gfx1250.
// GEMMs use v_wmma_f32_16x16x32_f16 (f16 operands, f32 accumulate), one wave
// per 16x64 output tile: a single A fragment feeds 4 WMMAs per K-step.
// Edge softmax/aggregation are flat per-edge kernels hitting the 192MB L2.
// Pre-BN biases cancel mathematically (affine-free batch norm) and are skipped.
// ---------------------------------------------------------------------------

typedef __attribute__((ext_vector_type(16))) _Float16 v16h;
typedef __attribute__((ext_vector_type(8)))  float    v8f;

#define N_NODES 50000
#define N_EDGES 800000
#define F_IN    128
#define HID     64
#define HEADS   4
#define NEG_SLOPE 0.2f
#define EPS_BN  1e-5f

// ------------------------------- utilities ---------------------------------

__global__ void fill_u32_kernel(unsigned* __restrict__ p, unsigned v, long long n) {
    long long i = (long long)blockIdx.x * blockDim.x + threadIdx.x;
    if (i < n) p[i] = v;
}

// monotone float <-> ordered-uint transform (for atomic max on floats)
__device__ __forceinline__ unsigned f2ord(float f) {
    unsigned u = __float_as_uint(f);
    return (u & 0x80000000u) ? ~u : (u | 0x80000000u);
}
__device__ __forceinline__ float ord2f(unsigned k) {
    unsigned u = (k & 0x80000000u) ? (k & 0x7fffffffu) : ~k;
    return __uint_as_float(u);
}

// ------------------------- weight pack (f32 -> f16^T) ----------------------
// W is [K, C] row-major fp32; produce BT [C, K] row-major f16 (i.e. W^T).
__global__ void pack_wT_kernel(const float* __restrict__ W, _Float16* __restrict__ BT,
                               int K, int C) {
    int idx = blockIdx.x * blockDim.x + threadIdx.x;
    if (idx >= K * C) return;
    int k = idx / C, c = idx - k * C;
    BT[(long long)c * K + k] = (_Float16)W[idx];
}

// ------------------------------ WMMA GEMM ----------------------------------
// Cout[M, C] = A[M, K] (fp32, cast to f16) x B (BT = [C, K] f16).
// One wave per 16x64 output tile: one A fragment, 4 B fragments, 4 WMMAs per
// 32-deep K-step. K must be a multiple of 32; M of 16; C of 64.
__global__ void gemm_wmma_kernel(const float* __restrict__ A,
                                 const _Float16* __restrict__ BT,
                                 float* __restrict__ Cout,
                                 int K, int C) {
    const int lane = threadIdx.x;            // 0..31, wave32
    const int r    = lane & 15;              // row (A) / col (B) within tile
    const int hs   = lane >> 4;              // half-select
    const long long row = (long long)blockIdx.x * 16 + r;  // node row
    const int col0 = blockIdx.y * 64;        // output-column tile base

    v8f acc[4];
#pragma unroll
    for (int t = 0; t < 4; ++t)
#pragma unroll
        for (int i = 0; i < 8; ++i) acc[t][i] = 0.0f;

    const float*    arowb = A + row * K + hs * 16;
    const _Float16* browb[4];
#pragma unroll
    for (int t = 0; t < 4; ++t)
        browb[t] = BT + (long long)(col0 + t * 16 + r) * K + hs * 16;

    for (int k0 = 0; k0 < K; k0 += 32) {
        // A fragment: 16x32 f16. lane holds row r; halves j = K=k0+hs*16+j.
        v16h a;
        const float* arow = arowb + k0;
#pragma unroll
        for (int j = 0; j < 8; ++j) {
            a[j]     = (_Float16)arow[j];
            a[j + 8] = (_Float16)arow[8 + j];
        }
        // B fragments: 32x16 f16. lane holds column r; halves j = K=k0+hs*16+j.
#pragma unroll
        for (int t = 0; t < 4; ++t) {
            v16h b;
            const _Float16* brow = browb[t] + k0;
#pragma unroll
            for (int j = 0; j < 16; ++j) b[j] = brow[j];
            acc[t] = __builtin_amdgcn_wmma_f32_16x16x32_f16(
                false, a, false, b, (short)0, acc[t], false, false);
        }
    }

    // C/D layout: acc[t][i] -> M = i + 8*hs, N = r (+ t*16 column offset)
    float* crow = Cout + ((long long)blockIdx.x * 16 + hs * 8) * C + col0 + r;
#pragma unroll
    for (int t = 0; t < 4; ++t)
#pragma unroll
        for (int i = 0; i < 8; ++i) crow[(long long)i * C + t * 16] = acc[t][i];
}

// --------------------------- attention scalars -----------------------------
// a_src[n,h] = sum_j h[n,h,j]*att_s[h,j] ; same for a_dst. h is [N, H*64].
__global__ void att_coef_kernel(const float* __restrict__ h,
                                const float* __restrict__ att_s,
                                const float* __restrict__ att_d,
                                float* __restrict__ a_s, float* __restrict__ a_d,
                                int H) {
    int idx = blockIdx.x * blockDim.x + threadIdx.x;
    if (idx >= N_NODES * H) return;
    int n = idx / H, hd = idx - n * H;
    const float* hp = h + ((long long)n * H + hd) * HID;
    const float* ws = att_s + hd * HID;
    const float* wd = att_d + hd * HID;
    float ss = 0.f, sd = 0.f;
#pragma unroll 8
    for (int j = 0; j < HID; ++j) {
        float v = hp[j];
        ss += v * ws[j];
        sd += v * wd[j];
    }
    a_s[idx] = ss;
    a_d[idx] = sd;
}

// ----------------------------- edge kernels --------------------------------
__device__ __forceinline__ void edge_endpoints(const int* __restrict__ ei,
                                               int e, int& s, int& d) {
    if (e < N_EDGES) { s = ei[e]; d = ei[N_EDGES + e]; }
    else             { s = d = e - N_EDGES; }           // appended self-loops
}

__global__ void edge_max_kernel(const int* __restrict__ ei,
                                const float* __restrict__ a_s,
                                const float* __restrict__ a_d,
                                unsigned* __restrict__ mkey, int H) {
    int idx = blockIdx.x * blockDim.x + threadIdx.x;
    int total = (N_EDGES + N_NODES) * H;
    if (idx >= total) return;
    int e = idx / H, hd = idx - e * H;
    int s, d; edge_endpoints(ei, e, s, d);
    float v = a_s[s * H + hd] + a_d[d * H + hd];
    v = v > 0.f ? v : NEG_SLOPE * v;                    // leaky_relu
    atomicMax(&mkey[d * H + hd], f2ord(v));
}

__global__ void edge_expsum_kernel(const int* __restrict__ ei,
                                   const float* __restrict__ a_s,
                                   const float* __restrict__ a_d,
                                   const unsigned* __restrict__ mkey,
                                   float* __restrict__ wbuf,
                                   float* __restrict__ denom, int H) {
    int idx = blockIdx.x * blockDim.x + threadIdx.x;
    int total = (N_EDGES + N_NODES) * H;
    if (idx >= total) return;
    int e = idx / H, hd = idx - e * H;
    int s, d; edge_endpoints(ei, e, s, d);
    float v = a_s[s * H + hd] + a_d[d * H + hd];
    v = v > 0.f ? v : NEG_SLOPE * v;
    float w = expf(v - ord2f(mkey[d * H + hd]));
    wbuf[idx] = w;
    atomicAdd(&denom[d * H + hd], w);
}

__global__ void edge_norm_kernel(const int* __restrict__ ei,
                                 const float* __restrict__ denom,
                                 float* __restrict__ wbuf, int H) {
    int idx = blockIdx.x * blockDim.x + threadIdx.x;
    int total = (N_EDGES + N_NODES) * H;
    if (idx >= total) return;
    int e = idx / H, hd = idx - e * H;
    int s, d; edge_endpoints(ei, e, s, d);
    (void)s;
    wbuf[idx] = wbuf[idx] / (denom[d * H + hd] + 1e-16f);
}

// out[dst, c] += alpha[e, c/64] * h[src, c]  — consecutive threads cover
// consecutive columns of one edge => coalesced gather + coalesced atomics.
__global__ void edge_aggregate_kernel(const int* __restrict__ ei,
                                      const float* __restrict__ alpha,
                                      const float* __restrict__ h,
                                      float* __restrict__ out,
                                      int H, int lC /* log2(H*64) */) {
    long long idx = (long long)blockIdx.x * blockDim.x + threadIdx.x;
    const int C = 1 << lC;
    long long total = (long long)(N_EDGES + N_NODES) << lC;
    if (idx >= total) return;
    int e = (int)(idx >> lC);
    int c = (int)(idx & (C - 1));
    int hd = c >> 6;                                    // HID == 64
    int s, d; edge_endpoints(ei, e, s, d);
    float a = alpha[(long long)e * H + hd];
    atomicAdd(&out[(long long)d * C + c], h[(long long)s * C + c] * a);
}

// ------------------------------ batch norm ---------------------------------
// sums[0:C]=sum, sums[C:2C]=sum of squares. blockDim.x == C.
__global__ void bn_stats_kernel(const float* __restrict__ x,
                                float* __restrict__ sums, int C) {
    int c = threadIdx.x;
    float s = 0.f, s2 = 0.f;
    for (int r = blockIdx.x; r < N_NODES; r += gridDim.x) {
        float v = x[(long long)r * C + c];
        s += v; s2 += v * v;
    }
    atomicAdd(&sums[c], s);
    atomicAdd(&sums[C + c], s2);
}

__global__ void bn_apply_relu_kernel(float* __restrict__ x,
                                     const float* __restrict__ sums, int C) {
    long long idx = (long long)blockIdx.x * blockDim.x + threadIdx.x;
    long long total = (long long)N_NODES * C;
    if (idx >= total) return;
    int c = (int)(idx % C);
    const float inv_n = 1.0f / (float)N_NODES;
    float mu  = sums[c] * inv_n;
    float var = sums[C + c] * inv_n - mu * mu;          // biased variance
    float y = (x[idx] - mu) * rsqrtf(var + EPS_BN);
    x[idx] = fmaxf(y, 0.f);
}

// ------------------------------ final head ---------------------------------
__global__ void final_proj_kernel(const float* __restrict__ x,
                                  const float* __restrict__ lw2,
                                  const float* __restrict__ lb2,
                                  float* __restrict__ out) {
    int n = blockIdx.x * blockDim.x + threadIdx.x;
    if (n >= N_NODES) return;
    float a0 = lb2[0], a1 = lb2[1];
    const float* xr = x + (long long)n * HID;
#pragma unroll 8
    for (int j = 0; j < HID; ++j) {
        float v = xr[j];
        a0 += v * lw2[j * 2 + 0];
        a1 += v * lw2[j * 2 + 1];
    }
    out[n * 2 + 0] = a0;
    out[n * 2 + 1] = a1;
}

// ------------------------------- host side ---------------------------------

static inline int cdiv(long long a, long long b) { return (int)((a + b - 1) / b); }

struct Scratch {
    float*    bufH;   // [N, 256]
    float*    bufO;   // [N, 256]
    float*    aS;     // [N, HEADS]
    float*    aD;     // [N, HEADS]
    unsigned* mkey;   // [N, HEADS]
    float*    denom;  // [N, HEADS]
    float*    wAl;    // [E+N, HEADS]
    _Float16* wpk;    // up to 256*256 halves
    float*    sums;   // [2*256]
};

static void run_gat_layer(const float* xin, int K, int H,
                          const float* W, const float* atts, const float* attd,
                          const Scratch& sc, float* hbuf, float* obuf,
                          const int* ei, hipStream_t stream) {
    const int C  = H * HID;
    const int lC = (C == 256) ? 8 : 6;
    const int EH = (N_EDGES + N_NODES) * H;

    // pack W^T to f16
    pack_wT_kernel<<<cdiv((long long)K * C, 256), 256, 0, stream>>>(W, sc.wpk, K, C);
    // h = x @ W  (WMMA)
    dim3 gg(N_NODES / 16, C / 64);
    gemm_wmma_kernel<<<gg, 32, 0, stream>>>(xin, sc.wpk, hbuf, K, C);
    // per-node attention scalars
    att_coef_kernel<<<cdiv((long long)N_NODES * H, 256), 256, 0, stream>>>(
        hbuf, atts, attd, sc.aS, sc.aD, H);
    // init: max-keys (0 == lowest key), denom, output accumulator
    fill_u32_kernel<<<cdiv((long long)N_NODES * H, 256), 256, 0, stream>>>(
        sc.mkey, 0u, (long long)N_NODES * H);
    fill_u32_kernel<<<cdiv((long long)N_NODES * H, 256), 256, 0, stream>>>(
        (unsigned*)sc.denom, 0u, (long long)N_NODES * H);
    fill_u32_kernel<<<cdiv((long long)N_NODES * C, 256), 256, 0, stream>>>(
        (unsigned*)obuf, 0u, (long long)N_NODES * C);
    // segment softmax over incoming edges
    edge_max_kernel<<<cdiv(EH, 256), 256, 0, stream>>>(ei, sc.aS, sc.aD, sc.mkey, H);
    edge_expsum_kernel<<<cdiv(EH, 256), 256, 0, stream>>>(
        ei, sc.aS, sc.aD, sc.mkey, sc.wAl, sc.denom, H);
    edge_norm_kernel<<<cdiv(EH, 256), 256, 0, stream>>>(ei, sc.denom, sc.wAl, H);
    // message aggregation
    long long tot = (long long)(N_EDGES + N_NODES) << lC;
    edge_aggregate_kernel<<<cdiv(tot, 256), 256, 0, stream>>>(
        ei, sc.wAl, hbuf, obuf, H, lC);
}

static void run_bn_relu(float* x, int C, const Scratch& sc, hipStream_t stream) {
    fill_u32_kernel<<<1, 2 * C, 0, stream>>>((unsigned*)sc.sums, 0u, 2 * C);
    bn_stats_kernel<<<256, C, 0, stream>>>(x, sc.sums, C);
    bn_apply_relu_kernel<<<cdiv((long long)N_NODES * C, 256), 256, 0, stream>>>(
        x, sc.sums, C);
}

extern "C" void kernel_launch(void* const* d_in, const int* in_sizes, int n_in,
                              void* d_out, int out_size, void* d_ws, size_t ws_size,
                              hipStream_t stream) {
    const float* x   = (const float*)d_in[0];
    const int*   ei  = (const int*)d_in[1];
    const float* W0  = (const float*)d_in[2];
    const float* as0 = (const float*)d_in[3];
    const float* ad0 = (const float*)d_in[4];
    const float* W1  = (const float*)d_in[6];
    const float* as1 = (const float*)d_in[7];
    const float* ad1 = (const float*)d_in[8];
    const float* W2  = (const float*)d_in[10];
    const float* as2 = (const float*)d_in[11];
    const float* ad2 = (const float*)d_in[12];
    const float* lw1 = (const float*)d_in[14];
    const float* lw2 = (const float*)d_in[16];
    const float* lb2 = (const float*)d_in[17];
    float* out = (float*)d_out;
    (void)in_sizes; (void)n_in; (void)out_size; (void)ws_size;

    // carve workspace
    char* p = (char*)d_ws;
    auto take = [&](size_t bytes) {
        void* r = (void*)p;
        p += (bytes + 255) & ~(size_t)255;
        return r;
    };
    Scratch sc;
    sc.bufH  = (float*)   take((size_t)N_NODES * 256 * 4);
    sc.bufO  = (float*)   take((size_t)N_NODES * 256 * 4);
    sc.aS    = (float*)   take((size_t)N_NODES * HEADS * 4);
    sc.aD    = (float*)   take((size_t)N_NODES * HEADS * 4);
    sc.mkey  = (unsigned*)take((size_t)N_NODES * HEADS * 4);
    sc.denom = (float*)   take((size_t)N_NODES * HEADS * 4);
    sc.wAl   = (float*)   take((size_t)(N_EDGES + N_NODES) * HEADS * 4);
    sc.wpk   = (_Float16*)take((size_t)256 * 256 * 2);
    sc.sums  = (float*)   take((size_t)2 * 256 * 4);

    // ---- layer 0: F_IN=128 -> 4 heads x 64 (concat) ----
    run_gat_layer(x, F_IN, HEADS, W0, as0, ad0, sc, sc.bufH, sc.bufO, ei, stream);
    run_bn_relu(sc.bufO, HEADS * HID, sc, stream);

    // ---- layer 1: 256 -> 64, heads=1 ----  (bufO is input; bufH reused)
    run_gat_layer(sc.bufO, HEADS * HID, 1, W1, as1, ad1, sc, sc.bufH, sc.bufO, ei, stream);
    run_bn_relu(sc.bufO, HID, sc, stream);

    // ---- layer 2: 64 -> 64, heads=1 ----
    run_gat_layer(sc.bufO, HID, 1, W2, as2, ad2, sc, sc.bufH, sc.bufO, ei, stream);
    run_bn_relu(sc.bufO, HID, sc, stream);

    // ---- MLP head: (x @ lw1) -> BN -> relu -> @ lw2 + lb2 ----
    pack_wT_kernel<<<cdiv((long long)HID * HID, 256), 256, 0, stream>>>(
        lw1, sc.wpk, HID, HID);
    dim3 gg(N_NODES / 16, HID / 64);
    gemm_wmma_kernel<<<gg, 32, 0, stream>>>(sc.bufO, sc.wpk, sc.bufH, HID, HID);
    run_bn_relu(sc.bufH, HID, sc, stream);
    final_proj_kernel<<<cdiv(N_NODES, 256), 256, 0, stream>>>(sc.bufH, lw2, lb2, out);
}
